// GraphFlow_EGNN_22471268892736
// MI455X (gfx1250) — matrix operand order repaired and checked by reading, not compiled, not run on hardware
//
#include <hip/hip_runtime.h>

// ---------------------------------------------------------------------------
// EGNN (4 layers, H=32) fused WMMA implementation for gfx1250 (wave32).
// - All GEMMs use v_wmma_f32_16x16x32_f16 with f32 accumulation.
// - Edge-feature gathers use GLOBAL_LOAD_ASYNC_TO_LDS_B128 (ASYNCcnt path).
// ---------------------------------------------------------------------------

#define HDIM  32
#define WSTR  104   // halves per transposed-weight row (K padded to 96, 208B rows, 16B aligned)
#define ESTR  104   // edge-input LDS row stride (halves)
#define ASTR  40    // activation LDS row stride (halves) : 80B, 16B aligned, bank-friendly
#define NSTR  72    // node-input LDS row stride (halves) : 144B

typedef __attribute__((ext_vector_type(16))) _Float16     f16x16;
typedef __attribute__((ext_vector_type(8)))  float        f32x8;
typedef __attribute__((ext_vector_type(4)))  unsigned int u32x4;

__device__ __forceinline__ void lds_fence() {
  // CDNA5 split counters: drain outstanding DS ops so cross-lane LDS writes are visible.
  asm volatile("s_wait_dscnt 0" ::: "memory");
}

__device__ __forceinline__ void async_fence() {
  // Drain outstanding async global->LDS transfers (ASYNCcnt).
  asm volatile("s_wait_asynccnt 0" ::: "memory");
}

// Async copy of 32 bytes from global memory directly into LDS (per-lane addresses).
// IOFFSET is added to both the LDS and global address per the CDNA5 ISA.
__device__ __forceinline__ void async_gather32(unsigned lds_addr, unsigned long long gaddr) {
  asm volatile("global_load_async_to_lds_b128 %0, %1, off"
               :: "v"(lds_addr), "v"(gaddr) : "memory");
  asm volatile("global_load_async_to_lds_b128 %0, %1, off offset:16"
               :: "v"(lds_addr), "v"(gaddr) : "memory");
}

__device__ __forceinline__ float silu(float x) { return x / (1.0f + __expf(-x)); }

__device__ __forceinline__ f32x8 wmma16(f16x16 a, f16x16 b, f32x8 c) {
  // D = A(16x32 f16) * B(32x16 f16) + C(16x16 f32)
  return __builtin_amdgcn_wmma_f32_16x16x32_f16(false, a, false, b, (short)0, c, false, false);
}

// A fragment (16 rows x 32 K, f16) from LDS row-major [row][K] with strideH halves.
// Lane L holds row = L%16, K = kb + (L<16?0:8)+[0..7] and +16 (two ds_load_b128).
__device__ __forceinline__ f16x16 load_a_frag(const _Float16* base, int strideH, int kb, int lane) {
  int r    = lane & 15;
  int koff = kb + ((lane & 16) ? 8 : 0);
  const char* p = (const char*)(base + r * strideH + koff);
  union { f16x16 h; u32x4 u[2]; } t;
  t.u[0] = *(const u32x4*)(p);
  t.u[1] = *(const u32x4*)(p + 32);
  return t.h;
}

// B fragment (32 K x 16 N, f16) from transposed weights [N][K] with strideH halves.
// Lane L holds col N = n0 + L%16, K = kb + (L<16?0:8) + [0..7] and +16 (two 16B loads).
__device__ __forceinline__ f16x16 load_b_frag(const _Float16* base, int strideH, int kb, int lane, int n0) {
  int r    = n0 + (lane & 15);
  int koff = kb + ((lane & 16) ? 8 : 0);
  const char* p = (const char*)(base + r * strideH + koff);
  union { f16x16 h; u32x4 u[2]; } t;
  t.u[0] = *(const u32x4*)(p);
  t.u[1] = *(const u32x4*)(p + 32);
  return t.h;
}

// ------------------------------ helper kernels ------------------------------

__global__ void k_transpose_f16(const float* __restrict__ src, _Float16* __restrict__ dst,
                                int K, int O) {
  int i = blockIdx.x * blockDim.x + threadIdx.x;
  if (i >= O * WSTR) return;
  int o = i / WSTR, k = i - o * WSTR;
  dst[i] = (k < K) ? (_Float16)src[k * O + o] : (_Float16)0.0f;
}

__global__ void k_zero_f32(float* p, int n) {
  int i = blockIdx.x * blockDim.x + threadIdx.x;
  if (i < n) p[i] = 0.0f;
}

__global__ void k_copy_f32(const float* __restrict__ s, float* __restrict__ d, int n) {
  int i = blockIdx.x * blockDim.x + threadIdx.x;
  if (i < n) d[i] = s[i];
}

__global__ void k_count(const int* __restrict__ row, float* cnt, int n) {
  int i = blockIdx.x * blockDim.x + threadIdx.x;
  if (i < n) atomicAdd(&cnt[row[i]], 1.0f);
}

__global__ void k_initx(const float* __restrict__ pos, float* cnt, float* x, int n) {
  int i = blockIdx.x * blockDim.x + threadIdx.x;
  if (i < n) {
    cnt[i] = fmaxf(cnt[i], 1.0f);
    x[i*3+0] = pos[i*2+0];
    x[i*3+1] = pos[i*2+1];
    x[i*3+2] = 0.0f;
  }
}

// ------------------------------ embedding -----------------------------------
// h = [t | data] @ emb_in_w + b   ([N,65] @ [65,32])

struct EmbWave { _Float16 ein[16 * ESTR]; };
struct EmbLDS  { _Float16 w[32 * WSTR]; float bias[HDIM]; EmbWave wv[4]; };

__global__ void __launch_bounds__(128) egnn_embed_kernel(
    const float* __restrict__ data, const float* __restrict__ tval,
    const _Float16* __restrict__ wT, const float* __restrict__ bvec,
    float* __restrict__ h, _Float16* __restrict__ hf, int numTiles)
{
  __shared__ __align__(16) EmbLDS s;
  int tid = threadIdx.x, lane = tid & 31, wid = tid >> 5;
  {
    const u32x4* srcv = (const u32x4*)wT;
    u32x4* dstv = (u32x4*)s.w;
    for (int i = tid; i < 32 * WSTR * 2 / 16; i += 128) dstv[i] = srcv[i];
    if (tid < HDIM) s.bias[tid] = bvec[tid];
  }
  __syncthreads();
  float t0 = tval[0];
  int nWaves = (gridDim.x * blockDim.x) >> 5;
  int gw = blockIdx.x * 4 + wid;
  int el = lane & 15, part = lane >> 4, n_lo = lane & 15, mbase = (lane & 16) ? 8 : 0;
  for (int tile = gw; tile < numTiles; tile += nWaves) {
    _Float16* ein = s.wv[wid].ein;
    int node = tile * 16 + el;
    const float* drow = data + (size_t)node * 64;
    _Float16* erow = ein + el * ESTR;
    if (part == 0) erow[0] = (_Float16)t0;
    for (int i = 0; i < 32; i++) erow[1 + part * 32 + i] = (_Float16)drow[part * 32 + i];
    if (part == 1) { for (int i = 0; i < 31; i++) erow[65 + i] = (_Float16)0.0f; }
    lds_fence();
    f32x8 a0 = {}, a1 = {};
#pragma unroll
    for (int kb = 0; kb < 96; kb += 32) {
      f16x16 A = load_a_frag(ein, ESTR, kb, lane);
      a0 = wmma16(A, load_b_frag(s.w, WSTR, kb, lane, 0),  a0);
      a1 = wmma16(A, load_b_frag(s.w, WSTR, kb, lane, 16), a1);
    }
    float b0 = s.bias[n_lo], b1 = s.bias[16 + n_lo];
#pragma unroll
    for (int r = 0; r < 8; r++) {
      int nd = tile * 16 + mbase + r;
      float v0 = a0[r] + b0, v1 = a1[r] + b1;
      size_t i0 = (size_t)nd * HDIM + n_lo, i1 = i0 + 16;
      h[i0] = v0; h[i1] = v1;
      hf[i0] = (_Float16)v0; hf[i1] = (_Float16)v1;
    }
    lds_fence();
  }
}

// ------------------------------ fused edge kernel ----------------------------
// Per 16-edge tile: async gather -> edge MLP (66->32->32, silu) -> agg scatter
//                   -> coord MLP (32->32 silu ->1) -> phi -> x scatter.

struct EdgeWave {
  _Float16 ein[16 * ESTR];
  _Float16 act1[16 * ASTR];
  _Float16 act2[16 * ASTR];
  float    diff[16 * 4];
  float    phi[16];
  int      rowi[16];
};
struct EdgeLDS { _Float16 w[3 * 32 * WSTR]; float bias[4 * HDIM]; EdgeWave wv[4]; };

__global__ void __launch_bounds__(128) egnn_edge_kernel(
    const _Float16* __restrict__ hf, const float* __restrict__ xc,
    const float* __restrict__ ea, const int* __restrict__ row, const int* __restrict__ col,
    const _Float16* __restrict__ wT,
    const float* __restrict__ b1, const float* __restrict__ b2,
    const float* __restrict__ cb1, const float* __restrict__ cw2,
    const float* __restrict__ cnt,
    float* __restrict__ agg, float* __restrict__ xn, int numTiles)
{
  __shared__ __align__(16) EdgeLDS s;
  int tid = threadIdx.x, lane = tid & 31, wid = tid >> 5;
  {
    const u32x4* srcv = (const u32x4*)wT;
    u32x4* dstv = (u32x4*)s.w;
    for (int i = tid; i < 3 * 32 * WSTR * 2 / 16; i += 128) dstv[i] = srcv[i];
    if (tid < HDIM) {
      s.bias[tid]       = b1[tid];
      s.bias[32 + tid]  = b2[tid];
      s.bias[64 + tid]  = cb1[tid];
      s.bias[96 + tid]  = cw2[tid];
    }
  }
  __syncthreads();
  const _Float16* w1 = s.w;
  const _Float16* w2 = s.w + 32 * WSTR;
  const _Float16* c1 = s.w + 2 * 32 * WSTR;
  int nWaves = (gridDim.x * blockDim.x) >> 5;
  int gw = blockIdx.x * 4 + wid;
  int el = lane & 15, part = lane >> 4, n_lo = lane & 15, mbase = (lane & 16) ? 8 : 0;

  for (int tile = gw; tile < numTiles; tile += nWaves) {
    EdgeWave& W = s.wv[wid];
    int e = tile * 16 + el;
    int ri = row[e], ci = col[e];

    // Prefetch next tile's index/attr streams into cache (global_prefetch_b8).
    if (tile + nWaves < numTiles) {
      int en = (tile + nWaves) * 16 + el;
      __builtin_prefetch(&row[en], 0, 1);
      __builtin_prefetch(&col[en], 0, 1);
      __builtin_prefetch(&ea[en], 0, 1);
    }

    { // async gather of h[row] (ein halves 0..31) and h[col] (halves 32..63), 32B per lane
      unsigned ldsbase = (unsigned)(uintptr_t)(W.ein + el * ESTR) + (unsigned)(part * 32);
      unsigned long long gr = (unsigned long long)(uintptr_t)(hf + (size_t)ri * HDIM) + (unsigned)(part * 32);
      unsigned long long gc = (unsigned long long)(uintptr_t)(hf + (size_t)ci * HDIM) + (unsigned)(part * 32);
      async_gather32(ldsbase,       gr);
      async_gather32(ldsbase + 64,  gc);
    }
    if (part == 0) {
      float d0 = xc[ri * 3 + 0] - xc[ci * 3 + 0];
      float d1 = xc[ri * 3 + 1] - xc[ci * 3 + 1];
      float d2 = xc[ri * 3 + 2] - xc[ci * 3 + 2];
      W.diff[el * 4 + 0] = d0; W.diff[el * 4 + 1] = d1; W.diff[el * 4 + 2] = d2;
      float rad = d0 * d0 + d1 * d1 + d2 * d2;
      W.ein[el * ESTR + 64] = (_Float16)rad;
      W.ein[el * ESTR + 65] = (_Float16)ea[e];
      W.rowi[el] = ri;
    } else { // zero halves [66..103]
      char* rb = (char*)(W.ein + el * ESTR);
      *(unsigned int*)(rb + 132) = 0u;
      *(unsigned int*)(rb + 136) = 0u;
      *(unsigned int*)(rb + 140) = 0u;
      u32x4 z = (u32x4){0, 0, 0, 0};
      ((u32x4*)rb)[9] = z; ((u32x4*)rb)[10] = z; ((u32x4*)rb)[11] = z; ((u32x4*)rb)[12] = z;
    }
    async_fence();
    lds_fence();

    // edge MLP layer 1: [16x66(+pad)] @ [66x32]
    f32x8 a0 = {}, a1 = {};
#pragma unroll
    for (int kb = 0; kb < 96; kb += 32) {
      f16x16 A = load_a_frag(W.ein, ESTR, kb, lane);
      a0 = wmma16(A, load_b_frag(w1, WSTR, kb, lane, 0),  a0);
      a1 = wmma16(A, load_b_frag(w1, WSTR, kb, lane, 16), a1);
    }
    {
      float q0 = s.bias[n_lo], q1 = s.bias[16 + n_lo];
#pragma unroll
      for (int r = 0; r < 8; r++) {
        W.act1[(mbase + r) * ASTR + n_lo]      = (_Float16)silu(a0[r] + q0);
        W.act1[(mbase + r) * ASTR + 16 + n_lo] = (_Float16)silu(a1[r] + q1);
      }
    }
    lds_fence();

    // edge MLP layer 2: [16x32] @ [32x32] -> m, scatter to agg
    f32x8 m0 = {}, m1 = {};
    {
      f16x16 A = load_a_frag(W.act1, ASTR, 0, lane);
      m0 = wmma16(A, load_b_frag(w2, WSTR, 0, lane, 0),  m0);
      m1 = wmma16(A, load_b_frag(w2, WSTR, 0, lane, 16), m1);
    }
    {
      float q0 = s.bias[32 + n_lo], q1 = s.bias[48 + n_lo];
#pragma unroll
      for (int r = 0; r < 8; r++) {
        float v0 = silu(m0[r] + q0), v1 = silu(m1[r] + q1);
        m0[r] = v0; m1[r] = v1;
        W.act2[(mbase + r) * ASTR + n_lo]      = (_Float16)v0;
        W.act2[(mbase + r) * ASTR + 16 + n_lo] = (_Float16)v1;
      }
    }
#pragma unroll
    for (int r = 0; r < 8; r++) {
      int rr = W.rowi[mbase + r];
      atomicAdd(&agg[(size_t)rr * HDIM + n_lo],      m0[r]);
      atomicAdd(&agg[(size_t)rr * HDIM + 16 + n_lo], m1[r]);
    }
    lds_fence();

    // coord MLP: silu([16x32]@[32x32]+cb1) . cw2  -> phi per edge
    f32x8 h0 = {}, h1 = {};
    {
      f16x16 A = load_a_frag(W.act2, ASTR, 0, lane);
      h0 = wmma16(A, load_b_frag(c1, WSTR, 0, lane, 0),  h0);
      h1 = wmma16(A, load_b_frag(c1, WSTR, 0, lane, 16), h1);
    }
    {
      float q0 = s.bias[64 + n_lo], q1 = s.bias[80 + n_lo];
      float s0 = s.bias[96 + n_lo], s1 = s.bias[112 + n_lo];
#pragma unroll
      for (int r = 0; r < 8; r++) {
        float p = silu(h0[r] + q0) * s0 + silu(h1[r] + q1) * s1;
        p += __shfl_xor(p, 1);
        p += __shfl_xor(p, 2);
        p += __shfl_xor(p, 4);
        p += __shfl_xor(p, 8);
        if (n_lo == 0) W.phi[mbase + r] = p;
      }
    }
    lds_fence();
    if (lane < 16) { // equivariant coordinate scatter (mean agg folded per-edge)
      float p  = W.phi[lane];
      int   rr = W.rowi[lane];
      float ic = 1.0f / cnt[rr];
      atomicAdd(&xn[rr * 3 + 0], W.diff[lane * 4 + 0] * p * ic);
      atomicAdd(&xn[rr * 3 + 1], W.diff[lane * 4 + 1] * p * ic);
      atomicAdd(&xn[rr * 3 + 2], W.diff[lane * 4 + 2] * p * ic);
    }
    lds_fence();
  }
}

// ------------------------------ node update ---------------------------------
// h += silu([h|agg] @ nw1 + nb1) @ nw2 + nb2

struct NodeWave { _Float16 ein[16 * NSTR]; _Float16 act1[16 * ASTR]; };
struct NodeLDS  { _Float16 w[2 * 32 * WSTR]; float bias[2 * HDIM]; NodeWave wv[4]; };

__global__ void __launch_bounds__(128) egnn_node_kernel(
    float* __restrict__ h, _Float16* __restrict__ hf, const float* __restrict__ agg,
    const _Float16* __restrict__ wT, const float* __restrict__ nb1,
    const float* __restrict__ nb2, int numTiles)
{
  __shared__ __align__(16) NodeLDS s;
  int tid = threadIdx.x, lane = tid & 31, wid = tid >> 5;
  {
    const u32x4* srcv = (const u32x4*)wT;
    u32x4* dstv = (u32x4*)s.w;
    for (int i = tid; i < 2 * 32 * WSTR * 2 / 16; i += 128) dstv[i] = srcv[i];
    if (tid < HDIM) { s.bias[tid] = nb1[tid]; s.bias[32 + tid] = nb2[tid]; }
  }
  __syncthreads();
  const _Float16* w1 = s.w;
  const _Float16* w2 = s.w + 32 * WSTR;
  int nWaves = (gridDim.x * blockDim.x) >> 5;
  int gw = blockIdx.x * 4 + wid;
  int el = lane & 15, part = lane >> 4, n_lo = lane & 15, mbase = (lane & 16) ? 8 : 0;
  for (int tile = gw; tile < numTiles; tile += nWaves) {
    NodeWave& W = s.wv[wid];
    int node = tile * 16 + el;
    { // h row via async global->LDS; agg converted f32->f16 through VGPRs
      unsigned ldsb = (unsigned)(uintptr_t)(W.ein + el * NSTR) + (unsigned)(part * 32);
      unsigned long long gh = (unsigned long long)(uintptr_t)(hf + (size_t)node * HDIM) + (unsigned)(part * 32);
      async_gather32(ldsb, gh);
      const float* ar = agg + (size_t)node * HDIM + part * 16;
      _Float16* er = W.ein + el * NSTR + 32 + part * 16;
      for (int i = 0; i < 16; i++) er[i] = (_Float16)ar[i];
    }
    async_fence();
    lds_fence();
    f32x8 a0 = {}, a1 = {};
#pragma unroll
    for (int kb = 0; kb < 64; kb += 32) {
      f16x16 A = load_a_frag(W.ein, NSTR, kb, lane);
      a0 = wmma16(A, load_b_frag(w1, WSTR, kb, lane, 0),  a0);
      a1 = wmma16(A, load_b_frag(w1, WSTR, kb, lane, 16), a1);
    }
    {
      float q0 = s.bias[n_lo], q1 = s.bias[16 + n_lo];
#pragma unroll
      for (int r = 0; r < 8; r++) {
        W.act1[(mbase + r) * ASTR + n_lo]      = (_Float16)silu(a0[r] + q0);
        W.act1[(mbase + r) * ASTR + 16 + n_lo] = (_Float16)silu(a1[r] + q1);
      }
    }
    lds_fence();
    f32x8 u0 = {}, u1 = {};
    {
      f16x16 A = load_a_frag(W.act1, ASTR, 0, lane);
      u0 = wmma16(A, load_b_frag(w2, WSTR, 0, lane, 0),  u0);
      u1 = wmma16(A, load_b_frag(w2, WSTR, 0, lane, 16), u1);
    }
    {
      float q0 = s.bias[32 + n_lo], q1 = s.bias[48 + n_lo];
#pragma unroll
      for (int r = 0; r < 8; r++) {
        size_t i0 = (size_t)(tile * 16 + mbase + r) * HDIM + n_lo, i1 = i0 + 16;
        float v0 = h[i0] + u0[r] + q0;
        float v1 = h[i1] + u1[r] + q1;
        h[i0] = v0; h[i1] = v1;
        hf[i0] = (_Float16)v0; hf[i1] = (_Float16)v1;
      }
    }
    lds_fence();
  }
}

// ------------------------------ output proj ---------------------------------
// out = h @ emb_out_w + b    ([N,32] @ [32,64])

struct OutWave { _Float16 ein[16 * ASTR]; };
struct OutLDS  { _Float16 w[64 * WSTR]; float bias[64]; OutWave wv[4]; };

__global__ void __launch_bounds__(128) egnn_out_kernel(
    const _Float16* __restrict__ hf, const _Float16* __restrict__ wT,
    const float* __restrict__ bvec, float* __restrict__ out, int numTiles)
{
  __shared__ __align__(16) OutLDS s;
  int tid = threadIdx.x, lane = tid & 31, wid = tid >> 5;
  {
    const u32x4* srcv = (const u32x4*)wT;
    u32x4* dstv = (u32x4*)s.w;
    for (int i = tid; i < 64 * WSTR * 2 / 16; i += 128) dstv[i] = srcv[i];
    if (tid < 64) s.bias[tid] = bvec[tid];
  }
  __syncthreads();
  int nWaves = (gridDim.x * blockDim.x) >> 5;
  int gw = blockIdx.x * 4 + wid;
  int el = lane & 15, part = lane >> 4, n_lo = lane & 15, mbase = (lane & 16) ? 8 : 0;
  for (int tile = gw; tile < numTiles; tile += nWaves) {
    _Float16* ein = s.wv[wid].ein;
    int node = tile * 16 + el;
    {
      unsigned ldsb = (unsigned)(uintptr_t)(ein + el * ASTR) + (unsigned)(part * 32);
      unsigned long long gh = (unsigned long long)(uintptr_t)(hf + (size_t)node * HDIM) + (unsigned)(part * 32);
      async_gather32(ldsb, gh);
    }
    async_fence();
    lds_fence();
    f16x16 A = load_a_frag(ein, ASTR, 0, lane);
#pragma unroll
    for (int nt = 0; nt < 4; nt++) {
      f32x8 acc = {};
      acc = wmma16(A, load_b_frag(s.w, WSTR, 0, lane, nt * 16), acc);
      float bb = s.bias[nt * 16 + n_lo];
#pragma unroll
      for (int r = 0; r < 8; r++)
        out[(size_t)(tile * 16 + mbase + r) * 64 + nt * 16 + n_lo] = acc[r] + bb;
    }
    lds_fence();
  }
}

// ------------------------------ launcher ------------------------------------

extern "C" void kernel_launch(void* const* d_in, const int* in_sizes, int n_in,
                              void* d_out, int out_size, void* d_ws, size_t ws_size,
                              hipStream_t stream) {
  const float* data      = (const float*)d_in[0];
  const float* pos       = (const float*)d_in[1];
  const float* edge_attr = (const float*)d_in[2];
  const float* tval      = (const float*)d_in[3];
  const int*   row       = (const int*)d_in[4];
  const int*   col       = (const int*)d_in[5];
  const float* emb_in_w  = (const float*)d_in[6];
  const float* emb_in_b  = (const float*)d_in[7];
  const float* emb_out_w = (const float*)d_in[8];
  const float* emb_out_b = (const float*)d_in[9];
  const float* edge_w1   = (const float*)d_in[10];
  const float* edge_b1   = (const float*)d_in[11];
  const float* edge_w2   = (const float*)d_in[12];
  const float* edge_b2   = (const float*)d_in[13];
  const float* node_w1   = (const float*)d_in[14];
  const float* node_b1   = (const float*)d_in[15];
  const float* node_w2   = (const float*)d_in[16];
  const float* node_b2   = (const float*)d_in[17];
  const float* coord_w1  = (const float*)d_in[18];
  const float* coord_b1  = (const float*)d_in[19];
  const float* coord_w2  = (const float*)d_in[20];
  float* out = (float*)d_out;

  const int N = in_sizes[0] / 64;   // 50000
  const int E = in_sizes[4];        // 800000
  const int L = 4;
  const int tilesN = N / 16, tilesE = E / 16;  // exact multiples for this problem

  (void)n_in; (void)out_size; (void)ws_size;

  // workspace layout
  char* ws = (char*)d_ws;
  auto alloc = [&](size_t bytes) {
    char* p = ws; ws += (bytes + 255) & ~(size_t)255; return p;
  };
  float*    h   = (float*)   alloc((size_t)N * HDIM * 4);
  _Float16* hf  = (_Float16*)alloc((size_t)N * HDIM * 2);
  float*    x0  = (float*)   alloc((size_t)N * 3 * 4);
  float*    x1  = (float*)   alloc((size_t)N * 3 * 4);
  float*    agg = (float*)   alloc((size_t)N * HDIM * 4);
  float*    cnt = (float*)   alloc((size_t)N * 4);
  size_t wHalves = (size_t)32 * WSTR * (1 + 3 * L + 2 * L) + (size_t)64 * WSTR;
  _Float16* wbuf = (_Float16*)alloc(wHalves * 2);

  _Float16* embinT = wbuf;
  _Float16* edgeT  = embinT + 32 * WSTR;                 // per layer: [e1T | e2T | c1T]
  _Float16* nodeT  = edgeT + (size_t)L * 3 * 32 * WSTR;  // per layer: [nw1T | nw2T]
  _Float16* eoT    = nodeT + (size_t)L * 2 * 32 * WSTR;

  dim3 tb(256);
  auto blks = [](int n) { return dim3((unsigned)((n + 255) / 256)); };

  // weight transposes (f32 [K][O] -> f16 [O][WSTR])
  k_transpose_f16<<<blks(32 * WSTR), tb, 0, stream>>>(emb_in_w, embinT, 65, 32);
  for (int l = 0; l < L; l++) {
    _Float16* eb = edgeT + (size_t)l * 3 * 32 * WSTR;
    k_transpose_f16<<<blks(32 * WSTR), tb, 0, stream>>>(edge_w1 + (size_t)l * 66 * 32, eb, 66, 32);
    k_transpose_f16<<<blks(32 * WSTR), tb, 0, stream>>>(edge_w2 + (size_t)l * 32 * 32, eb + 32 * WSTR, 32, 32);
    k_transpose_f16<<<blks(32 * WSTR), tb, 0, stream>>>(coord_w1 + (size_t)l * 32 * 32, eb + 2 * 32 * WSTR, 32, 32);
    _Float16* nb = nodeT + (size_t)l * 2 * 32 * WSTR;
    k_transpose_f16<<<blks(32 * WSTR), tb, 0, stream>>>(node_w1 + (size_t)l * 64 * 32, nb, 64, 32);
    k_transpose_f16<<<blks(32 * WSTR), tb, 0, stream>>>(node_w2 + (size_t)l * 32 * 32, nb + 32 * WSTR, 32, 32);
  }
  k_transpose_f16<<<blks(64 * WSTR), tb, 0, stream>>>(emb_out_w, eoT, 32, 64);

  // degree counts (clamped) and padded coords
  k_zero_f32<<<blks(N), tb, 0, stream>>>(cnt, N);
  k_count<<<blks(E), tb, 0, stream>>>(row, cnt, E);
  k_initx<<<blks(N), tb, 0, stream>>>(pos, cnt, x0, N);

  // input embedding
  egnn_embed_kernel<<<dim3(512), dim3(128), 0, stream>>>(data, tval, embinT, emb_in_b, h, hf, tilesN);

  float* xc = x0;
  float* xn = x1;
  for (int l = 0; l < L; l++) {
    k_zero_f32<<<blks(N * HDIM), tb, 0, stream>>>(agg, N * HDIM);
    k_copy_f32<<<blks(N * 3), tb, 0, stream>>>(xc, xn, N * 3);
    egnn_edge_kernel<<<dim3(1024), dim3(128), 0, stream>>>(
        hf, xc, edge_attr, row, col,
        edgeT + (size_t)l * 3 * 32 * WSTR,
        edge_b1 + (size_t)l * 32, edge_b2 + (size_t)l * 32,
        coord_b1 + (size_t)l * 32, coord_w2 + (size_t)l * 32,
        cnt, agg, xn, tilesE);
    egnn_node_kernel<<<dim3(512), dim3(128), 0, stream>>>(
        h, hf, agg, nodeT + (size_t)l * 2 * 32 * WSTR,
        node_b1 + (size_t)l * 32, node_b2 + (size_t)l * 32, tilesN);
    float* tmp = xc; xc = xn; xn = tmp;
  }

  egnn_out_kernel<<<dim3(512), dim3(128), 0, stream>>>(hf, eoT, emb_out_b, out, tilesN);
}
